// DynamicGraphConstructor_9663676416330
// MI455X (gfx1250) — matrix operand order repaired and checked by reading, not compiled, not run on hardware
//
#include <hip/hip_runtime.h>
#include <hip/hip_bf16.h>
#include <math.h>

// ---------------- types ----------------
typedef __bf16 bf16;
typedef __attribute__((ext_vector_type(8)))  __bf16 bf16x8;
typedef __attribute__((ext_vector_type(16))) __bf16 v16bf;
typedef __attribute__((ext_vector_type(8)))  float  v8f;
typedef __attribute__((ext_vector_type(4)))  float  f32x4;
typedef __attribute__((ext_vector_type(4)))  unsigned int u32x4;
typedef __attribute__((ext_vector_type(4)))  int    i32x4;
typedef __attribute__((ext_vector_type(8)))  int    i32x8;

// ---------------- problem constants ----------------
#define BT    96     // B*T = 8*12
#define NROW  250    // nodes
#define NPAD  256    // padded to 16 WMMA tiles
#define DEMB  64     // embedding dim
#define KIN   224    // 128+64+32 concat features
#define KT    3      // temporal tiling factor
#define CH    (BT * NROW * NROW * KT)   // elements per output chunk

#define CAT16 0, 1, 2, 3, 4, 5, 6, 7, 8, 9, 10, 11, 12, 13, 14, 15

// ------------------------------------------------------------
// Fragment builders for GLOBAL operands (per ISA 7.12.2, wave32).
// ------------------------------------------------------------
__device__ __forceinline__ v16bf load_fragA(const bf16* __restrict__ row,
                                            int k0, int half) {
  bf16x8 lo = *(const bf16x8*)(row + k0 + 8 * half);
  bf16x8 hi = *(const bf16x8*)(row + k0 + 16 + 8 * half);
  return __builtin_shufflevector(lo, hi, CAT16);
}

__device__ __forceinline__ v16bf load_fragB(const bf16* __restrict__ col,
                                            int k0, int half) {
  bf16x8 lo = *(const bf16x8*)(col + k0 + 8 * half);
  bf16x8 hi = *(const bf16x8*)(col + k0 + 16 + 8 * half);
  return __builtin_shufflevector(lo, hi, CAT16);
}

__device__ __forceinline__ v16bf load_fragA_f32(const float* __restrict__ seg,
                                                int half) {
  const float* p0 = seg + 8 * half;
  const float* p1 = seg + 16 + 8 * half;
  f32x4 q0 = *(const f32x4*)(p0);
  f32x4 q1 = *(const f32x4*)(p0 + 4);
  f32x4 q2 = *(const f32x4*)(p1);
  f32x4 q3 = *(const f32x4*)(p1 + 4);
  v16bf a;
#pragma unroll
  for (int i = 0; i < 4; ++i) {
    a[i]      = (bf16)q0[i];
    a[4 + i]  = (bf16)q1[i];
    a[8 + i]  = (bf16)q2[i];
    a[12 + i] = (bf16)q3[i];
  }
  return a;
}

// ------------------------------------------------------------
// TDM: DMA a 2-D bf16 panel (rows x rowlen, row stride in elements)
// from global memory into LDS at byte offset lds_off.
// D# packing per ISA 8.3/8.4.
// ------------------------------------------------------------
__device__ __forceinline__ void tdm_load_panel_2d(const void* gsrc,
                                                  unsigned lds_off,
                                                  unsigned rows,
                                                  unsigned rowlen,
                                                  unsigned row_stride) {
  const unsigned long long ga = (unsigned long long)(uintptr_t)gsrc;
  u32x4 g0;
  g0[0] = 1u;                                       // count=1 (valid user D#)
  g0[1] = lds_off;                                  // lds_addr (bytes)
  g0[2] = (unsigned)ga;                             // global_addr[31:0]
  g0[3] = (unsigned)((ga >> 32) & 0x1FFFFFFu)       // global_addr[56:32]
        | (2u << 30);                               // type = 2 ("image")
  i32x8 g1;
  g1[0] = (int)(1u << 16);                          // data_size = 1 -> 2 bytes
  g1[1] = (int)((rowlen & 0xFFFFu) << 16);          // tensor_dim0[15:0]
  g1[2] = (int)(((rowlen >> 16) & 0xFFFFu)          // tensor_dim0[31:16]
        | ((rows & 0xFFFFu) << 16));                // tensor_dim1[15:0]
  g1[3] = (int)(((rows >> 16) & 0xFFFFu)            // tensor_dim1[31:16]
        | ((rowlen & 0xFFFFu) << 16));              // tile_dim0
  g1[4] = (int)(rows & 0xFFFFu);                    // tile_dim1 (tile_dim2=0)
  g1[5] = (int)row_stride;                          // tensor_dim0_stride[31:0]
  g1[6] = 0;
  g1[7] = 0;
  const i32x4 z4 = {0, 0, 0, 0};
#if __clang_major__ >= 23
  const i32x8 z8 = {0, 0, 0, 0, 0, 0, 0, 0};
  __builtin_amdgcn_tensor_load_to_lds(g0, g1, z4, z4, z8, 0);
#else
  __builtin_amdgcn_tensor_load_to_lds(g0, g1, z4, z4, 0);
#endif
}

// ============================================================
// K0: transpose W_emb [224,64] fp32 -> wembT [64,224] bf16 (tiny)
// ============================================================
__global__ __launch_bounds__(64) void prep_wemb(const float* __restrict__ wemb,
                                                bf16* __restrict__ wembT) {
  const int k = blockIdx.x;
  const int n = threadIdx.x;
  wembT[(size_t)n * KIN + k] = (bf16)wemb[(size_t)k * DEMB + n];
}

// ============================================================
// K1: nodevec = tanh(concat(hist,prior,obs) @ W_emb + b_emb)
// ============================================================
__global__ __launch_bounds__(32) void embed_kernel(
    const float* __restrict__ hist, const float* __restrict__ prior,
    const float* __restrict__ obs,  const bf16* __restrict__ wembT,
    const float* __restrict__ bemb, bf16* __restrict__ nvb)
{
  const int mtile = blockIdx.x;
  const int bt    = blockIdx.y;
  const int lane  = threadIdx.x;
  const int half  = lane >> 4;
  const int lm    = lane & 15;

  const int arow = mtile * 16 + lm;
  const int rowc = (arow < NROW) ? arow : (NROW - 1);
  const float* xh = hist  + ((size_t)bt * NROW + rowc) * 128;
  const float* xp = prior + ((size_t)bt * NROW + rowc) * 64;
  const float* xo = obs   + ((size_t)bt * NROW + rowc) * 32;

  v8f c0 = {}, c1 = {}, c2 = {}, c3 = {};
#pragma unroll
  for (int k0 = 0; k0 < KIN; k0 += 32) {
    const float* seg = (k0 < 128) ? (xh + k0)
                     : (k0 < 192) ? (xp + (k0 - 128))
                                  : (xo + (k0 - 192));
    const v16bf a = load_fragA_f32(seg, half);
    const v16bf b0 = load_fragB(wembT + (size_t)(0 * 16 + lm) * KIN, k0, half);
    const v16bf b1 = load_fragB(wembT + (size_t)(1 * 16 + lm) * KIN, k0, half);
    const v16bf b2 = load_fragB(wembT + (size_t)(2 * 16 + lm) * KIN, k0, half);
    const v16bf b3 = load_fragB(wembT + (size_t)(3 * 16 + lm) * KIN, k0, half);
    c0 = __builtin_amdgcn_wmma_f32_16x16x32_bf16(false, a, false, b0, (short)0, c0, false, false);
    c1 = __builtin_amdgcn_wmma_f32_16x16x32_bf16(false, a, false, b1, (short)0, c1, false, false);
    c2 = __builtin_amdgcn_wmma_f32_16x16x32_bf16(false, a, false, b2, (short)0, c2, false, false);
    c3 = __builtin_amdgcn_wmma_f32_16x16x32_bf16(false, a, false, b3, (short)0, c3, false, false);
  }

  v8f acc[4] = {c0, c1, c2, c3};
#pragma unroll
  for (int s = 0; s < 4; ++s) {
    const int ncol = s * 16 + lm;
    const float bias = bemb[ncol];
#pragma unroll
    for (int i = 0; i < 8; ++i) {
      const int row = mtile * 16 + i + half * 8;
      const float val = (row < NROW) ? tanhf(acc[s][i] + bias) : 0.f;
      nvb[((size_t)bt * NPAD + row) * DEMB + ncol] = (bf16)val;
    }
  }
}

// ============================================================
// K2: sim = relu(nodevec @ nodevec^T) -> fp32 [NPAD x NPAD]
// ============================================================
__global__ __launch_bounds__(32) void sim_kernel(
    const bf16* __restrict__ nvb, float* __restrict__ sim)
{
  const int nquad = blockIdx.x;
  const int mtile = blockIdx.y;
  const int bt    = blockIdx.z;
  const int lane  = threadIdx.x;
  const int half  = lane >> 4;
  const int lm    = lane & 15;

  const bf16* base = nvb + (size_t)bt * NPAD * DEMB;
  const bf16* arowp = base + (size_t)(mtile * 16 + lm) * DEMB;
  const int nbase = nquad * 64;

  v8f c0 = {}, c1 = {}, c2 = {}, c3 = {};
#pragma unroll
  for (int k0 = 0; k0 < DEMB; k0 += 32) {
    const v16bf a = load_fragA(arowp, k0, half);
    const v16bf b0 = load_fragB(base + (size_t)(nbase + 0 * 16 + lm) * DEMB, k0, half);
    const v16bf b1 = load_fragB(base + (size_t)(nbase + 1 * 16 + lm) * DEMB, k0, half);
    const v16bf b2 = load_fragB(base + (size_t)(nbase + 2 * 16 + lm) * DEMB, k0, half);
    const v16bf b3 = load_fragB(base + (size_t)(nbase + 3 * 16 + lm) * DEMB, k0, half);
    c0 = __builtin_amdgcn_wmma_f32_16x16x32_bf16(false, a, false, b0, (short)0, c0, false, false);
    c1 = __builtin_amdgcn_wmma_f32_16x16x32_bf16(false, a, false, b1, (short)0, c1, false, false);
    c2 = __builtin_amdgcn_wmma_f32_16x16x32_bf16(false, a, false, b2, (short)0, c2, false, false);
    c3 = __builtin_amdgcn_wmma_f32_16x16x32_bf16(false, a, false, b3, (short)0, c3, false, false);
  }

  v8f acc[4] = {c0, c1, c2, c3};
#pragma unroll
  for (int s = 0; s < 4; ++s) {
    const int col = nbase + s * 16 + lm;
#pragma unroll
    for (int i = 0; i < 8; ++i) {
      const int row = mtile * 16 + i + half * 8;
      sim[((size_t)bt * NPAD + row) * NPAD + col] = fmaxf(acc[s][i], 0.f);
    }
  }
}

// ============================================================
// K3: normalization vectors (sim symmetric -> P1 == P2):
//   d[n]  = (rowsum(sim)[n]+1e-9)^-1/2
//   cm[m] = d[m] / (d[m]*sum_n sim[m,n]*d[n] + 1e-9)
// ============================================================
__global__ __launch_bounds__(256) void norm_kernel(
    const float* __restrict__ sim, float* __restrict__ dvec,
    float* __restrict__ cmvec)
{
  const int bt = blockIdx.x;
  const int n  = threadIdx.x;
  __shared__ float sd[NPAD];

  const f32x4* srow = (const f32x4*)(sim + ((size_t)bt * NPAD + n) * NPAD);
  float s = 0.f;
#pragma unroll 4
  for (int q = 0; q < NPAD / 4; ++q) {
    const f32x4 v = srow[q];
    s += v.x + v.y + v.z + v.w;
  }
  const float d = 1.0f / sqrtf(s + 1e-9f);
  sd[n] = d;
  dvec[bt * NPAD + n] = d;
  __syncthreads();

  float t = 0.f;
#pragma unroll 4
  for (int q = 0; q < NPAD / 4; ++q) {
    const f32x4 v = srow[q];
    t += v.x * sd[4 * q] + v.y * sd[4 * q + 1] +
         v.z * sd[4 * q + 2] + v.w * sd[4 * q + 3];
  }
  cmvec[bt * NPAD + n] = d / (d * t + 1e-9f);
}

// ============================================================
// K4: P[n,m] = d[n]*cm[m]*sim[n,m]; bf16 P and P^T (LDS transpose),
//     masked K_T-tiled copies to chunks 0/2 (NT stores).
// ============================================================
__global__ __launch_bounds__(256) void pwrite_kernel(
    const float* __restrict__ sim, const float* __restrict__ dvec,
    const float* __restrict__ cmvec, bf16* __restrict__ Pb,
    bf16* __restrict__ PbT, float* __restrict__ out)
{
  __shared__ float tile[16][17];
  const int bx = blockIdx.x;
  const int by = blockIdx.y;
  const int bt = blockIdx.z;
  const int tx = threadIdx.x;
  const int ty = threadIdx.y;
  const int n = by * 16 + ty;
  const int m = bx * 16 + tx;

  const float val = dvec[bt * NPAD + n] * cmvec[bt * NPAD + m] *
                    sim[((size_t)bt * NPAD + n) * NPAD + m];
  Pb[((size_t)bt * NPAD + n) * NPAD + m] = (bf16)val;
  tile[ty][tx] = val;
  __syncthreads();
  PbT[((size_t)bt * NPAD + bx * 16 + ty) * NPAD + by * 16 + tx] =
      (bf16)tile[tx][ty];

  if (n < NROW && m < NROW) {
    const float ov = (n == m) ? 0.f : val;
    const size_t base = ((size_t)(bt * NROW + n)) * (KT * NROW) + m;
#pragma unroll
    for (int k = 0; k < KT; ++k) {
      __builtin_nontemporal_store(ov, out + 0 * (size_t)CH + base + k * NROW);
      __builtin_nontemporal_store(ov, out + 2 * (size_t)CH + base + k * NROW);
    }
  }
}

// ============================================================
// K5: Q = P @ P. Block = 4 waves = 64x64 output tile.
//     TDM DMAs the 64x256 A panel (Pb) and 64x256 B panel (PbT)
//     into LDS; block syncs on s_wait_tensorcnt + barrier; each
//     wave then runs 8 K-steps x 4 col-tiles of bf16 WMMA fed by
//     ds_load_b128. The empty asm with the escaped LDS pointer +
//     "memory" clobber tells the compiler the TDM wrote the LDS
//     buffer (the builtin only receives an integer byte offset,
//     so without it the loads fold to poison and disappear).
// ============================================================
__global__ __launch_bounds__(128) void square_kernel(
    const bf16* __restrict__ Pb, const bf16* __restrict__ PbT,
    float* __restrict__ out)
{
  // [0] = A panel (64 rows x 256 k), [1] = B panel; bf16x8 granules.
  __shared__ bf16x8 lds[2][64][NPAD / 8];          // 64 KB total
  const int nquad = blockIdx.x;                    // 0..3 (64 output cols)
  const int mquad = blockIdx.y;                    // 0..3 (64 output rows)
  const int bt    = blockIdx.z;
  const int tid   = threadIdx.x;
  const int w     = tid >> 5;                      // wave id 0..3
  const int lane  = tid & 31;
  const int half  = lane >> 4;
  const int lm    = lane & 15;

  if (w == 0) {
    tdm_load_panel_2d(Pb  + ((size_t)bt * NPAD + mquad * 64) * NPAD,
                      /*lds_off=*/0u, /*rows=*/64u, /*rowlen=*/NPAD,
                      /*row_stride=*/NPAD);
    tdm_load_panel_2d(PbT + ((size_t)bt * NPAD + nquad * 64) * NPAD,
                      /*lds_off=*/(unsigned)sizeof(lds[0]),
                      /*rows=*/64u, /*rowlen=*/NPAD, /*row_stride=*/NPAD);
    __builtin_amdgcn_s_wait_tensorcnt(0);
  }
  __syncthreads();
  // Escape the LDS buffer so the TDM's writes are visible to alias
  // analysis; without this every ds_load below folds to poison.
  {
    void* esc = (void*)lds;
    asm volatile("" : "+r"(esc)::"memory");
  }

  const int ar = w * 16 + lm;                      // A row within panel
  v8f c0 = {}, c1 = {}, c2 = {}, c3 = {};
#pragma unroll
  for (int k0 = 0; k0 < NPAD; k0 += 32) {
    const int g  = k0 >> 3;                        // granule base (k0/8)
    const v16bf a = __builtin_shufflevector(lds[0][ar][g + 2 * half],
                                            lds[0][ar][g + 2 * half + 1], CAT16);
    const int gb = g + 2 * half;
    const v16bf b0 = __builtin_shufflevector(lds[1][0 * 16 + lm][gb],
                                             lds[1][0 * 16 + lm][gb + 1], CAT16);
    const v16bf b1 = __builtin_shufflevector(lds[1][1 * 16 + lm][gb],
                                             lds[1][1 * 16 + lm][gb + 1], CAT16);
    const v16bf b2 = __builtin_shufflevector(lds[1][2 * 16 + lm][gb],
                                             lds[1][2 * 16 + lm][gb + 1], CAT16);
    const v16bf b3 = __builtin_shufflevector(lds[1][3 * 16 + lm][gb],
                                             lds[1][3 * 16 + lm][gb + 1], CAT16);
    c0 = __builtin_amdgcn_wmma_f32_16x16x32_bf16(false, a, false, b0, (short)0, c0, false, false);
    c1 = __builtin_amdgcn_wmma_f32_16x16x32_bf16(false, a, false, b1, (short)0, c1, false, false);
    c2 = __builtin_amdgcn_wmma_f32_16x16x32_bf16(false, a, false, b2, (short)0, c2, false, false);
    c3 = __builtin_amdgcn_wmma_f32_16x16x32_bf16(false, a, false, b3, (short)0, c3, false, false);
  }

  v8f acc[4] = {c0, c1, c2, c3};
#pragma unroll
  for (int s = 0; s < 4; ++s) {
    const int col = nquad * 64 + s * 16 + lm;
    if (col >= NROW) continue;
#pragma unroll
    for (int i = 0; i < 8; ++i) {
      const int row = mquad * 64 + w * 16 + i + half * 8;
      if (row < NROW) {
        const float ov = (row == col) ? 0.f : acc[s][i];
        const size_t base = ((size_t)(bt * NROW + row)) * (KT * NROW) + col;
#pragma unroll
        for (int k = 0; k < KT; ++k) {
          __builtin_nontemporal_store(ov, out + 1 * (size_t)CH + base + k * NROW);
          __builtin_nontemporal_store(ov, out + 3 * (size_t)CH + base + k * NROW);
        }
      }
    }
  }
}

// ============================================================
// host-side launcher
// ============================================================
extern "C" void kernel_launch(void* const* d_in, const int* in_sizes, int n_in,
                              void* d_out, int out_size, void* d_ws, size_t ws_size,
                              hipStream_t stream) {
  const float* hist  = (const float*)d_in[0];  // [8,12,250,128]
  const float* prior = (const float*)d_in[1];  // [8,12,250,64]
  const float* obs   = (const float*)d_in[2];  // [8,12,250,32]
  const float* wemb  = (const float*)d_in[3];  // [224,64]
  const float* bemb  = (const float*)d_in[4];  // [64]
  float* out = (float*)d_out;                  // 4 x [8,12,250,750]

  char* ws = (char*)d_ws;
  const size_t szPb  = (size_t)BT * NPAD * NPAD * sizeof(bf16);
  const size_t szSim = (size_t)BT * NPAD * NPAD * sizeof(float);
  const size_t szNv  = (size_t)BT * NPAD * DEMB * sizeof(bf16);
  const size_t szVec = (size_t)BT * NPAD * sizeof(float);
  bf16*  Pb    = (bf16*)(ws);
  bf16*  PbT   = (bf16*)(ws + szPb);
  float* sim   = (float*)(ws + 2 * szPb);
  bf16*  nvb   = (bf16*)(ws + 2 * szPb + szSim);
  float* dvec  = (float*)(ws + 2 * szPb + szSim + szNv);
  float* cmvec = (float*)(ws + 2 * szPb + szSim + szNv + szVec);
  bf16*  wembT = (bf16*)(ws + 2 * szPb + szSim + szNv + 2 * szVec);

  prep_wemb<<<dim3(KIN), 64, 0, stream>>>(wemb, wembT);

  embed_kernel<<<dim3(NPAD / 16, BT), 32, 0, stream>>>(
      hist, prior, obs, wembT, bemb, nvb);

  sim_kernel<<<dim3(NPAD / 64, NPAD / 16, BT), 32, 0, stream>>>(nvb, sim);

  norm_kernel<<<dim3(BT), 256, 0, stream>>>(sim, dvec, cmvec);

  pwrite_kernel<<<dim3(NPAD / 16, NPAD / 16, BT), dim3(16, 16), 0, stream>>>(
      sim, dvec, cmvec, Pb, PbT, out);

  square_kernel<<<dim3(NPAD / 64, NPAD / 64, BT), 128, 0, stream>>>(
      Pb, PbT, out);
}
